// _IPEXAttention_90099823935702
// MI455X (gfx1250) — matrix operand order repaired and checked by reading, not compiled, not run on hardware
//
#include <hip/hip_runtime.h>
#include <hip/hip_bf16.h>
#include <stdint.h>

typedef __attribute__((ext_vector_type(16))) __bf16 v16bf;
typedef __attribute__((ext_vector_type(8)))  __bf16 v8bf;
typedef __attribute__((ext_vector_type(4)))  __bf16 v4bf;
typedef __attribute__((ext_vector_type(8)))  float  v8f;

union ABF { v16bf v; v8bf h[2]; };

#define NH   16
#define HD   128
#define HID  2048
#define SQ   1024
#define PAST 1024
#define KVL  2048
#define NB   4

#define USE_ASYNC 1
#if USE_ASYNC && defined(__has_builtin)
#if __has_builtin(__builtin_amdgcn_global_load_async_to_lds_b128)
#define HAVE_ASYNC 1
#endif
#endif
#ifndef HAVE_ASYNC
#define HAVE_ASYNC 0
#endif

static __device__ __forceinline__ v8f wmma_bf16(v16bf a, v16bf b, v8f c) {
  return __builtin_amdgcn_wmma_f32_16x16x32_bf16(false, a, false, b, (short)0, c,
                                                 false, false);
}

#if HAVE_ASYNC
// Builtin signature (from hipcc diagnostic): param 1 is a pointer to
// int __attribute__((vector_size(16))) in the global address space; param 2
// is the LDS-side pointer of the same pointee type.
typedef int v4i_vs __attribute__((vector_size(16)));
typedef __attribute__((address_space(1))) v4i_vs* as1_v4i_p;
typedef __attribute__((address_space(3))) v4i_vs* as3_v4i_p;

static __device__ __forceinline__ void async_b128(const void* g, void* l) {
  __builtin_amdgcn_global_load_async_to_lds_b128(
      (as1_v4i_p)(uintptr_t)g, (as3_v4i_p)(uintptr_t)l, 0, 0);
}
static __device__ __forceinline__ void wait_async_le16() {
  asm volatile("s_wait_asynccnt 0x10" ::: "memory");
}
static __device__ __forceinline__ void wait_async_0() {
  asm volatile("s_wait_asynccnt 0x0" ::: "memory");
}
#endif

// ---------------------------------------------------------------------------
// Kernel 1: fused QKV projection (bf16 WMMA, f32 accum) + in-register RoPE.
// 256 threads / 8 waves; block tile 128x128; wave strip 16(M)x128(N) so the
// rotate-half partner (d +/- 64) is accumulator j+/-4 in the same lane.
// Q -> Qw [B][NH][S][HD]; K(roped) -> Kc rows PAST..PAST+S; V -> Vrow [s][d].
// ---------------------------------------------------------------------------
__global__ __launch_bounds__(256) void qkv_rope_kernel(
    const float* __restrict__ X,
    const float* __restrict__ Wq, const float* __restrict__ Wk,
    const float* __restrict__ Wv,
    __bf16* __restrict__ Qw, __bf16* __restrict__ Kc, __bf16* __restrict__ Vrow)
{
  const int mat = blockIdx.z;
  const float* W = (mat == 0) ? Wq : (mat == 1) ? Wk : Wv;
  const int m0 = blockIdx.x * 128;
  const int hb = blockIdx.y;       // head index == 128-wide N block
  const int n0 = hb * 128;

  __shared__ alignas(16) __bf16 As[128][40];   // [m][k], 80B rows
  __shared__ alignas(16) __bf16 Bs[128][40];   // [n][k] = W transposed

  const int tid  = threadIdx.x;
  const int wave = tid >> 5;
  const int lane = tid & 31;
  const int kh   = lane >> 4;
  const int ln   = lane & 15;

  v8f acc[8];
#pragma unroll
  for (int j = 0; j < 8; ++j) acc[j] = (v8f)0.0f;

  for (int k0 = 0; k0 < HID; k0 += 32) {
    __syncthreads();
#pragma unroll
    for (int r = 0; r < 4; ++r) {
      int idx = tid + r * 256;
      {  // A tile: coalesced float4 reads, packed b64 LDS stores
        int row = idx >> 3, c4 = (idx & 7) * 4;
        float4 f = *(const float4*)(X + (size_t)(m0 + row) * HID + k0 + c4);
        v4bf p; p[0] = (__bf16)f.x; p[1] = (__bf16)f.y;
        p[2] = (__bf16)f.z; p[3] = (__bf16)f.w;
        *(v4bf*)&As[row][c4] = p;
      }
      {  // W^T tile: 4 coalesced dword reads down k, one packed b64 store
        int n = idx & 127, kg = idx >> 7;
        const float* wp = W + (size_t)(k0 + kg * 4) * HID + n0 + n;
        v4bf p; p[0] = (__bf16)wp[0];       p[1] = (__bf16)wp[HID];
        p[2] = (__bf16)wp[2 * HID]; p[3] = (__bf16)wp[3 * HID];
        *(v4bf*)&Bs[n][kg * 4] = p;
      }
    }
    __syncthreads();

    ABF a;
    a.h[0] = *(const v8bf*)&As[wave * 16 + ln][kh * 8];
    a.h[1] = *(const v8bf*)&As[wave * 16 + ln][kh * 8 + 16];
#pragma unroll
    for (int j = 0; j < 8; ++j) {
      ABF bb;
      bb.h[0] = *(const v8bf*)&Bs[j * 16 + ln][kh * 8];
      bb.h[1] = *(const v8bf*)&Bs[j * 16 + ln][kh * 8 + 16];
      acc[j] = wmma_bf16(a.v, bb.v, acc[j]);
    }
  }

  // RoPE frequencies: only 4 distinct per lane (d & 63 = (j&3)*16 + ln)
  float fr4[4];
#pragma unroll
  for (int jj = 0; jj < 4; ++jj)
    fr4[jj] = exp2f(-(float)(jj * 16 + ln) * (13.287712379549449f / 64.0f));

#pragma unroll
  for (int i = 0; i < 8; ++i) {
    int m = m0 + wave * 16 + i + 8 * kh;
    int s = m & (SQ - 1);
    size_t bh = (size_t)(m >> 10) * NH + hb;
    if (mat == 0 || mat == 1) {
      float pos = (float)(PAST + s);
      size_t base = (mat == 0) ? (bh * SQ + s) * HD
                               : (bh * KVL + PAST + s) * HD;
      __bf16* Out = (mat == 0) ? Qw : Kc;
#pragma unroll
      for (int j = 0; j < 8; ++j) {
        float sn, cs;
        __sincosf(pos * fr4[j & 3], &sn, &cs);
        float x = acc[j][i];
        float rot = (j < 4) ? -acc[j + 4][i] : acc[j - 4][i];
        Out[base + j * 16 + ln] = (__bf16)(x * cs + rot * sn);
      }
    } else {
      size_t base = (bh * SQ + s) * HD;
#pragma unroll
      for (int j = 0; j < 8; ++j) Vrow[base + j * 16 + ln] = (__bf16)acc[j][i];
    }
  }
}

// ---------------------------------------------------------------------------
// Kernel 0 (runs after kernel 1): pack KV cache to bf16.
//   Kc[t<PAST]   <- past_key (f32->bf16, coalesced, packed stores)
//   Vt[d][t]     <- transpose of past_value (f32) / Vrow (bf16) via LDS,
//                   fully coalesced global reads AND writes.
// ---------------------------------------------------------------------------
__global__ __launch_bounds__(128) void kv_pack_kernel(
    const float* __restrict__ Kp, const float* __restrict__ Vp,
    const __bf16* __restrict__ Vrow,
    __bf16* __restrict__ Kc, __bf16* __restrict__ Vt)
{
  const int t0 = blockIdx.x * 64;
  const size_t bh = (size_t)blockIdx.z * NH + blockIdx.y;
  const int tid = threadIdx.x;

  __shared__ alignas(16) __bf16 tile[128][72];   // [d][t]

  if (t0 < PAST) {
#pragma unroll
    for (int r = 0; r < 16; ++r) {
      int idx = tid + r * 128;
      int row = idx >> 5, c4 = (idx & 31) * 4;
      float4 f = *(const float4*)(Kp + (bh * PAST + t0 + row) * HD + c4);
      v4bf p; p[0] = (__bf16)f.x; p[1] = (__bf16)f.y;
      p[2] = (__bf16)f.z; p[3] = (__bf16)f.w;
      *(v4bf*)(Kc + (bh * KVL + t0 + row) * HD + c4) = p;
      float4 g = *(const float4*)(Vp + (bh * PAST + t0 + row) * HD + c4);
      tile[c4 + 0][row] = (__bf16)g.x; tile[c4 + 1][row] = (__bf16)g.y;
      tile[c4 + 2][row] = (__bf16)g.z; tile[c4 + 3][row] = (__bf16)g.w;
    }
  } else {
#pragma unroll
    for (int r = 0; r < 8; ++r) {
      int idx = tid + r * 128;
      int row = idx >> 4, c8 = (idx & 15) * 8;
      v8bf f = *(const v8bf*)(Vrow + (bh * SQ + (t0 - PAST) + row) * HD + c8);
#pragma unroll
      for (int e = 0; e < 8; ++e) tile[c8 + e][row] = f[e];
    }
  }
  __syncthreads();
#pragma unroll
  for (int r = 0; r < 8; ++r) {
    int idx = tid + r * 128;
    int d = idx >> 3, c8 = (idx & 7) * 8;
    *(v8bf*)(Vt + (bh * HD + d) * KVL + t0 + c8) = *(const v8bf*)&tile[d][c8];
  }
}

// ---------------------------------------------------------------------------
// Kernel 2: flash attention over uniform bf16 KV cache, double-buffered LDS,
// async global->LDS copies (GLOBAL_LOAD_ASYNC_TO_LDS_B128 / ASYNCcnt) when
// the builtin is available, synchronous fallback otherwise.
// ---------------------------------------------------------------------------
__shared__ alignas(16) __bf16 g_Ks[2][64][136];
__shared__ alignas(16) __bf16 g_Vs[2][128][72];
__shared__ alignas(16) __bf16 g_Ps[4][16][72];

static __device__ __forceinline__ void issue_tile(
    const __bf16* __restrict__ Kc, const __bf16* __restrict__ Vt,
    size_t bh, int t0, int buf, int tid)
{
#pragma unroll
  for (int r = 0; r < 8; ++r) {
    int idx = tid + r * 128;
    int row = idx >> 4, c8 = (idx & 15) * 8;           // K: 64 x 128
    const __bf16* gk = Kc + (bh * KVL + t0 + row) * HD + c8;
    int d = idx >> 3, c8v = (idx & 7) * 8;             // V: 128 x 64 (t-major)
    const __bf16* gv = Vt + (bh * HD + d) * KVL + t0 + c8v;
#if HAVE_ASYNC
    async_b128(gk, &g_Ks[buf][row][c8]);
    async_b128(gv, &g_Vs[buf][d][c8v]);
#else
    *(v8bf*)&g_Ks[buf][row][c8] = *(const v8bf*)gk;
    *(v8bf*)&g_Vs[buf][d][c8v] = *(const v8bf*)gv;
#endif
  }
}

__global__ __launch_bounds__(128) void attn_kernel(
    const __bf16* __restrict__ Q, const __bf16* __restrict__ Kc,
    const __bf16* __restrict__ Vt,
    const float* __restrict__ mask, float* __restrict__ Out)
{
  const int sb = blockIdx.x, h = blockIdx.y, b = blockIdx.z;
  const int tid = threadIdx.x, wave = tid >> 5, lane = tid & 31;
  const int kh = lane >> 4, ln = lane & 15;
  const size_t bh = (size_t)b * NH + h;

  // Preload roped-Q A fragments (K dim 128 -> 4 frags of 16x32)
  const __bf16* qptr = Q + (bh * SQ + sb * 64 + wave * 16 + ln) * HD;
  v16bf qa[4];
#pragma unroll
  for (int kk = 0; kk < 4; ++kk) {
    ABF u;
    u.h[0] = *(const v8bf*)(qptr + kk * 32 + kh * 8);
    u.h[1] = *(const v8bf*)(qptr + kk * 32 + kh * 8 + 16);
    qa[kk] = u.v;
  }

  v8f o[8];
#pragma unroll
  for (int j = 0; j < 8; ++j) o[j] = (v8f)0.0f;
  float rm[8], rl[8];
#pragma unroll
  for (int i = 0; i < 8; ++i) { rm[i] = -1e30f; rl[i] = 0.0f; }

  const float scale = 0.08838834764831845f;  // 1/sqrt(128)
  const int NT = KVL / 64;                   // 32 kv tiles

  issue_tile(Kc, Vt, bh, 0, 0, tid);
  issue_tile(Kc, Vt, bh, 64, 1, tid);

  for (int it = 0; it < NT; ++it) {
    const int t0 = it * 64;
    const int buf = it & 1;
#if HAVE_ASYNC
    if (it + 1 < NT) wait_async_le16(); else wait_async_0();
#endif
    __syncthreads();   // tile `it` resident in LDS for all waves

    // Scores: Q(16x32) x K^T(32x16), 4 N-subtiles x 4 K-steps
    v8f sc[4];
#pragma unroll
    for (int ns = 0; ns < 4; ++ns) {
      sc[ns] = (v8f)0.0f;
#pragma unroll
      for (int kk = 0; kk < 4; ++kk) {
        ABF u;
        u.h[0] = *(const v8bf*)&g_Ks[buf][ns * 16 + ln][kk * 32 + kh * 8];
        u.h[1] = *(const v8bf*)&g_Ks[buf][ns * 16 + ln][kk * 32 + kh * 8 + 16];
        sc[ns] = wmma_bf16(qa[kk], u.v, sc[ns]);
      }
    }

    // Online softmax (row spans the 16 lanes of this half-wave)
    const int srow_base = sb * 64 + wave * 16;
#pragma unroll
    for (int i = 0; i < 8; ++i) {
      int srow = srow_base + i + 8 * kh;
      const float* mrow = mask + ((size_t)b * SQ + srow) * KVL + t0;
      float cmax = -1e30f;
#pragma unroll
      for (int ns = 0; ns < 4; ++ns) {
        float v = sc[ns][i] * scale + mrow[ns * 16 + ln];
        sc[ns][i] = v;
        cmax = fmaxf(cmax, v);
      }
#pragma unroll
      for (int m = 1; m <= 8; m <<= 1)
        cmax = fmaxf(cmax, __shfl_xor(cmax, m, 32));
      float mnew = fmaxf(rm[i], cmax);
      float alpha = __expf(rm[i] - mnew);
      rm[i] = mnew;
      float psum = 0.0f;
#pragma unroll
      for (int ns = 0; ns < 4; ++ns) {
        float p = __expf(sc[ns][i] - mnew);
        psum += p;
        g_Ps[wave][i + 8 * kh][ns * 16 + ln] = (__bf16)p;
      }
#pragma unroll
      for (int m = 1; m <= 8; m <<= 1) psum += __shfl_xor(psum, m, 32);
      rl[i] = rl[i] * alpha + psum;
#pragma unroll
      for (int j = 0; j < 8; ++j) o[j][i] *= alpha;
    }

    // Re-read P as A fragments (same-wave DS ops are in order: no barrier)
    v16bf pa[2];
#pragma unroll
    for (int kk = 0; kk < 2; ++kk) {
      ABF u;
      u.h[0] = *(const v8bf*)&g_Ps[wave][ln][kk * 32 + kh * 8];
      u.h[1] = *(const v8bf*)&g_Ps[wave][ln][kk * 32 + kh * 8 + 16];
      pa[kk] = u.v;
    }
    // O += P(16x64) x V(64x128)
#pragma unroll
    for (int j = 0; j < 8; ++j) {
#pragma unroll
      for (int kk = 0; kk < 2; ++kk) {
        ABF u;
        u.h[0] = *(const v8bf*)&g_Vs[buf][j * 16 + ln][kk * 32 + kh * 8];
        u.h[1] = *(const v8bf*)&g_Vs[buf][j * 16 + ln][kk * 32 + kh * 8 + 16];
        o[j] = wmma_bf16(pa[kk], u.v, o[j]);
      }
    }
    __syncthreads();   // buffer free before it is refilled
    if (it + 2 < NT) issue_tile(Kc, Vt, bh, t0 + 128, buf, tid);
  }

  // Normalize and store fp32 [B][S][H]
#pragma unroll
  for (int i = 0; i < 8; ++i) {
    int srow = sb * 64 + wave * 16 + i + 8 * kh;
    float inv = 1.0f / rl[i];
    size_t base = ((size_t)b * SQ + srow) * HID + h * HD;
#pragma unroll
    for (int j = 0; j < 8; ++j)
      Out[base + j * 16 + ln] = o[j][i] * inv;
  }
}

extern "C" void kernel_launch(void* const* d_in, const int* in_sizes, int n_in,
                              void* d_out, int out_size, void* d_ws, size_t ws_size,
                              hipStream_t stream) {
  (void)in_sizes; (void)n_in; (void)out_size; (void)ws_size;
  const float* hs   = (const float*)d_in[0];
  const float* mask = (const float*)d_in[1];
  const float* kp   = (const float*)d_in[2];
  const float* vp   = (const float*)d_in[3];
  const float* Wq   = (const float*)d_in[4];
  const float* Wk   = (const float*)d_in[5];
  const float* Wv   = (const float*)d_in[6];
  float* out = (float*)d_out;

  const size_t QE = (size_t)NB * NH * SQ * HD;     // 8M elems
  const size_t KE = (size_t)NB * NH * KVL * HD;    // 16M elems
  __bf16* Qw   = (__bf16*)d_ws;
  __bf16* Kc   = Qw + QE;
  __bf16* Vt   = Kc + KE;
  __bf16* Vrow = Vt + KE;

  // 1) fused QKV GEMM + RoPE (Q, roped K into cache tail, V row-major)
  qkv_rope_kernel<<<dim3(32, 16, 3), 256, 0, stream>>>(hs, Wq, Wk, Wv, Qw, Kc, Vrow);
  // 2) pack past KV to bf16 + transpose all of V
  kv_pack_kernel<<<dim3(32, 16, 4), 128, 0, stream>>>(kp, vp, Vrow, Kc, Vt);
  // 3) flash attention, async double-buffered KV tiles
  attn_kernel<<<dim3(16, 16, 4), 128, 0, stream>>>(Qw, Kc, Vt, mask, out);
}
